// Kernel_90477781057668
// MI455X (gfx1250) — matrix operand order repaired and checked by reading, not compiled, not run on hardware
//
#include <hip/hip_runtime.h>
#include <hip/hip_bf16.h>
#include <math.h>

// ---------------------------------------------------------------------------
// Fused e3nn-style edge-kernel builder for MI455X (gfx1250, wave32, WMMA).
//   out[z, i, j] = sum_{y,w} M[i,j,y,w] * Y[z,y] * R[z,w]
//   R = MLP(radii), Y = real SH l=0..2, with self-interaction override.
// One workgroup (8 wave32) handles 128 edges, everything staged in LDS,
// all three GEMMs run on v_wmma_f32_16x16x32_f16 (f16 in, f32 accumulate).
// ---------------------------------------------------------------------------

typedef _Float16 half8  __attribute__((ext_vector_type(8)));
typedef _Float16 half16 __attribute__((ext_vector_type(16)));
typedef float    float8 __attribute__((ext_vector_type(8)));

#define TM      128      // edges per workgroup
#define HID     128      // MLP hidden width
#define NPATH   32
#define YDIM    9
#define KP      (YDIM * NPATH)   // 288 = contraction dim of mixing GEMM
#define NCOL    256              // DI*DJ output columns

// LDS strides in halves, padded (+8) so 16-lane fragment loads stripe banks.
#define STR_H   136      // for h1/h2/W2t/W3t (rows of 128)
#define STR_P   296      // for P / Mt        (rows of 288)

// LDS layout (byte offsets, all 16B aligned)
#define OFF_MT    0                       // 256*296*2 = 151552
#define OFF_W2T   151552                  // 128*136*2 =  34816
#define OFF_W3T   186368                  //  32*136*2 =   8704
#define OFF_P     195072                  // 128*296*2 =  75776
#define OFF_H1    OFF_P                   // aliases P (dead before P is built)
#define OFF_H2    (OFF_P + 34816)         // aliases P upper half
#define OFF_R     270848                  // 128*32*2  =   8192
#define OFF_Y     279040                  // 128*9*4   =   4608
#define OFF_RAD   283648                  // 128*4
#define OFF_NZ    284160                  // 128*4
#define OFF_SW1   284672                  // 128*4
#define OFF_SB1   285184                  // 128*4
#define OFF_SB2   285696                  // 128*4
#define OFF_SB3   286208                  //  32*4
#define OFF_SSW   286336                  //  32*4
#define LDS_BYTES 286720                  // 280 KB of the 320 KB WGP pool

// CDNA5 WMMA A-fragment: 16x32 f16. lane m=lane&15 owns row m; half-group
// h=lane>>4 owns K = {h*8..h*8+7, 16+h*8..16+h*8+7}  -> two 16B LDS loads.
__device__ __forceinline__ half16 frag_a(const _Float16* base, int row0,
                                         int k0, int stride, int lane) {
  const int m = lane & 15, hg = lane >> 4;
  const _Float16* p = base + (row0 + m) * stride + k0 + hg * 8;
  half16 f;
  *reinterpret_cast<half8*>(&f)        = *reinterpret_cast<const half8*>(p);
  *(reinterpret_cast<half8*>(&f) + 1)  = *reinterpret_cast<const half8*>(p + 16);
  return f;
}

// CDNA5 WMMA B-fragment: 32x16 f16. lane n=lane&15 owns column n; half-group
// owns K = {h*8..h*8+7, 16+h*8..16+h*8+7} of the TRANSPOSED image.
__device__ __forceinline__ half16 frag_b(const _Float16* base, int n0,
                                         int k0, int stride, int lane) {
  const int n = lane & 15, hg = lane >> 4;
  const _Float16* p = base + (n0 + n) * stride + k0 + hg * 8;
  half16 f;
  *reinterpret_cast<half8*>(&f)        = *reinterpret_cast<const half8*>(p);
  *(reinterpret_cast<half8*>(&f) + 1)  = *reinterpret_cast<const half8*>(p + 16);
  return f;
}

__device__ __forceinline__ float8 wmma_f16(half16 a, half16 b, float8 c) {
  return __builtin_amdgcn_wmma_f32_16x16x32_f16(false, a, false, b,
                                                (short)0, c, false, false);
}

__global__ void __launch_bounds__(256, 1)
sh_mixing_kernel(const float* __restrict__ r,
                 const float* __restrict__ Mg,    // [256][288] == [ij][yw] (!)
                 const float* __restrict__ W1g, const float* __restrict__ b1g,
                 const float* __restrict__ W2g, const float* __restrict__ b2g,
                 const float* __restrict__ W3g, const float* __restrict__ b3g,
                 const float* __restrict__ swg,
                 float* __restrict__ out, int nEdges) {
  extern __shared__ char smem[];
  _Float16* ldsM   = (_Float16*)(smem + OFF_MT);
  _Float16* ldsW2t = (_Float16*)(smem + OFF_W2T);
  _Float16* ldsW3t = (_Float16*)(smem + OFF_W3T);
  _Float16* ldsP   = (_Float16*)(smem + OFF_P);
  _Float16* ldsH1  = (_Float16*)(smem + OFF_H1);
  _Float16* ldsH2  = (_Float16*)(smem + OFF_H2);
  _Float16* ldsR   = (_Float16*)(smem + OFF_R);
  float*    Yv     = (float*)(smem + OFF_Y);
  float*    sRad   = (float*)(smem + OFF_RAD);
  float*    sNz    = (float*)(smem + OFF_NZ);
  float*    sW1    = (float*)(smem + OFF_SW1);
  float*    sB1    = (float*)(smem + OFF_SB1);
  float*    sB2    = (float*)(smem + OFF_SB2);
  float*    sB3    = (float*)(smem + OFF_SB3);
  float*    sSW    = (float*)(smem + OFF_SSW);

  const int tid  = threadIdx.x;
  const int lane = tid & 31;
  const int wv   = tid >> 5;
  const int blockBase = blockIdx.x * TM;

  // ---- Phase 0: stage weights into LDS as f16 (transposed where needed) ----
  // M[i,j,y,w] flat index = (ij)*288 + (yw): already B-fragment-transposed.
  for (int idx = tid; idx < NCOL * KP; idx += 256) {
    const int n = idx / KP, k = idx - n * KP;
    ldsM[n * STR_P + k] = (_Float16)Mg[idx];
  }
  for (int idx = tid; idx < HID * HID; idx += 256) {   // W2t[n][k] = W2[k][n]
    const int k = idx >> 7, n = idx & 127;
    ldsW2t[n * STR_H + k] = (_Float16)W2g[idx];
  }
  for (int idx = tid; idx < HID * NPATH; idx += 256) { // W3t[n][k] = W3[k][n]
    const int k = idx >> 5, n = idx & 31;
    ldsW3t[n * STR_H + k] = (_Float16)W3g[idx];
  }
  if (tid < 128) { sW1[tid] = W1g[tid]; sB1[tid] = b1g[tid]; sB2[tid] = b2g[tid]; }
  if (tid < 32)  { sB3[tid] = b3g[tid]; sSW[tid] = swg[tid]; }

  // ---- Phase 1: per-edge radii + spherical harmonics -----------------------
  if (tid < TM) {
    const int edge = blockBase + tid;
    float x = 0.f, y = 0.f, z = 0.f;
    if (edge < nEdges) { x = r[3*edge]; y = r[3*edge+1]; z = r[3*edge+2]; }
    const float sq  = x*x + y*y + z*z;
    const bool  nz  = sq > 0.f;
    const float rad = nz ? __builtin_sqrtf(sq) : 0.f;
    const float inv = nz ? 1.f / rad : 1.f;        // matches where(nz, radii, 1)
    const float dx = x*inv, dy = y*inv, dz = z*inv;
    const float C0  = 0.28209479177387814f, C1  = 0.4886025119029199f;
    const float C2A = 1.0925484305920792f,  C20 = 0.31539156525252005f;
    const float C22 = 0.5462742152960396f;
    float* yo = Yv + tid * YDIM;
    yo[0] = C0;
    yo[1] = C1 * dy;  yo[2] = C1 * dz;  yo[3] = C1 * dx;
    yo[4] = C2A * dx * dy;  yo[5] = C2A * dy * dz;
    yo[6] = C20 * (3.f * dz * dz - 1.f);
    yo[7] = C2A * dx * dz;  yo[8] = C22 * (dx * dx - dy * dy);
    sRad[tid] = rad;
    sNz[tid]  = nz ? 1.f : 0.f;
  }
  __syncthreads();

  // ---- Phase 2: h1 = relu(radii * W1 + b1), f16 into LDS -------------------
  for (int idx = tid; idx < TM * HID; idx += 256) {
    const int z = idx >> 7, k = idx & 127;
    const float hv = fmaf(sRad[z], sW1[k], sB1[k]);
    ldsH1[z * STR_H + k] = (_Float16)fmaxf(hv, 0.f);
  }
  __syncthreads();

  // ---- Phase 3: GEMM1 [128x128]@[128x128], relu -> h2 ----------------------
  {
    const int m0 = wv * 16;
    half16 a[4];
    #pragma unroll
    for (int kt = 0; kt < 4; ++kt) a[kt] = frag_a(ldsH1, m0, kt*32, STR_H, lane);
    for (int nt = 0; nt < 8; ++nt) {
      float8 acc = {};
      #pragma unroll
      for (int kt = 0; kt < 4; ++kt)
        acc = wmma_f16(a[kt], frag_b(ldsW2t, nt*16, kt*32, STR_H, lane), acc);
      const int n     = nt * 16 + (lane & 15);
      const int rbase = m0 + (lane >> 4) * 8;     // C/D: row = v + 8*(lane>=16)
      const float bias = sB2[n];
      #pragma unroll
      for (int v = 0; v < 8; ++v)
        ldsH2[(rbase + v) * STR_H + n] = (_Float16)fmaxf(acc[v] + bias, 0.f);
    }
  }
  __syncthreads();

  // ---- Phase 4: GEMM2 [128x128]@[128x32] + bias / self-weight -> R ---------
  {
    const int m0 = wv * 16;
    half16 a[4];
    #pragma unroll
    for (int kt = 0; kt < 4; ++kt) a[kt] = frag_a(ldsH2, m0, kt*32, STR_H, lane);
    for (int nt = 0; nt < 2; ++nt) {
      float8 acc = {};
      #pragma unroll
      for (int kt = 0; kt < 4; ++kt)
        acc = wmma_f16(a[kt], frag_b(ldsW3t, nt*16, kt*32, STR_H, lane), acc);
      const int n     = nt * 16 + (lane & 15);
      const int rbase = m0 + (lane >> 4) * 8;
      const float bias = sB3[n], sw = sSW[n];
      #pragma unroll
      for (int v = 0; v < 8; ++v) {
        const int row = rbase + v;
        float Rv = acc[v] + bias;
        if (sNz[row] == 0.f) Rv = sw;             // zero-radius: self-interaction
        ldsR[row * NPATH + n] = (_Float16)Rv;
      }
    }
  }
  __syncthreads();

  // ---- Phase 5: P[z, y*32+w] = Y[z,y] * R[z,w]  (overwrites h1/h2 alias) ---
  {
    const int z  = tid >> 1;
    const int c0 = (tid & 1) * (KP / 2);
    float yv[YDIM];
    #pragma unroll
    for (int y = 0; y < YDIM; ++y) yv[y] = Yv[z * YDIM + y];
    for (int c = c0; c < c0 + KP / 2; ++c) {
      const int y = c >> 5, w = c & 31;
      ldsP[z * STR_P + c] = (_Float16)(yv[y] * (float)ldsR[z * NPATH + w]);
    }
  }
  __syncthreads();

  // ---- Phase 6: GEMM3 [128x288]@[288x256] -> out (f32) ---------------------
  {
    const int m0 = wv * 16;
    half16 a[9];                                   // hoist A across 16 N-tiles
    #pragma unroll
    for (int kt = 0; kt < 9; ++kt) a[kt] = frag_a(ldsP, m0, kt*32, STR_P, lane);
    for (int nt = 0; nt < 16; ++nt) {
      float8 acc = {};
      #pragma unroll
      for (int kt = 0; kt < 9; ++kt)
        acc = wmma_f16(a[kt], frag_b(ldsM, nt*16, kt*32, STR_P, lane), acc);
      const int col   = nt * 16 + (lane & 15);
      const int rbase = m0 + (lane >> 4) * 8;
      #pragma unroll
      for (int v = 0; v < 8; ++v) {
        const int edge = blockBase + rbase + v;
        if (edge < nEdges) out[(size_t)edge * NCOL + col] = acc[v];
      }
    }
  }
}

extern "C" void kernel_launch(void* const* d_in, const int* in_sizes, int n_in,
                              void* d_out, int out_size, void* d_ws, size_t ws_size,
                              hipStream_t stream) {
  (void)n_in; (void)d_ws; (void)ws_size; (void)out_size;
  const float* r   = (const float*)d_in[0];
  const float* M   = (const float*)d_in[1];
  const float* W1  = (const float*)d_in[2];
  const float* b1  = (const float*)d_in[3];
  const float* W2  = (const float*)d_in[4];
  const float* b2  = (const float*)d_in[5];
  const float* W3  = (const float*)d_in[6];
  const float* b3  = (const float*)d_in[7];
  const float* sw  = (const float*)d_in[8];
  float* out = (float*)d_out;

  const int nEdges = in_sizes[0] / 3;
  const int blocks = (nEdges + TM - 1) / TM;

  (void)hipFuncSetAttribute((const void*)sh_mixing_kernel,
                            hipFuncAttributeMaxDynamicSharedMemorySize,
                            (int)LDS_BYTES);
  sh_mixing_kernel<<<blocks, 256, LDS_BYTES, stream>>>(
      r, M, W1, b1, W2, b2, W3, b3, sw, out, nEdges);
}
